// MSDeformAttn3D_89627377533663
// MI455X (gfx1250) — matrix operand order repaired and checked by reading, not compiled, not run on hardware
//
#include <hip/hip_runtime.h>
#include <stdint.h>

// ---------------- types ----------------
typedef __bf16  bf16x16 __attribute__((ext_vector_type(16)));
typedef float   f32x8   __attribute__((ext_vector_type(8)));
typedef int     i32x4   __attribute__((ext_vector_type(4)));

typedef __attribute__((address_space(1))) i32x4* gas_i32x4_ptr;   // global
typedef __attribute__((address_space(3))) i32x4* las_i32x4_ptr;   // LDS

// ---------------- problem constants ----------------
#define N_B     2
#define LQ      8192
#define DMODEL  256
#define HEADS   8
#define POINTS  4
#define NLVL    4
#define S_TOT   42240           // 32768 + 8192 + 1024 + 256
#define MQ      (N_B * LQ)      // 16384
#define MV      (N_B * S_TOT)   // 84480
#define KDIM    256

// ---------------- gfx1250 async global->LDS staging ----------------
#if defined(__has_builtin)
#  if __has_builtin(__builtin_amdgcn_global_load_async_to_lds_b128) && \
      __has_builtin(__builtin_amdgcn_s_wait_asynccnt)
#    define HAVE_ASYNC_LDS 1
#  endif
#endif

__device__ __forceinline__ void stage16(const __bf16* g, __bf16* l) {
#if defined(HAVE_ASYNC_LDS)
    __builtin_amdgcn_global_load_async_to_lds_b128(
        (gas_i32x4_ptr)g, (las_i32x4_ptr)l, 0, 0);
#else
    *(uint4*)l = *(const uint4*)g;
#endif
}

__device__ __forceinline__ void stage_wait() {
#if defined(HAVE_ASYNC_LDS)
    __builtin_amdgcn_s_wait_asynccnt(0);
#endif
}

// ---------------- f32 -> bf16 convert ----------------
__global__ void cvt_f32_bf16(const float* __restrict__ in, __bf16* __restrict__ out, size_t n) {
    size_t i = (size_t)blockIdx.x * blockDim.x + threadIdx.x;
    if (i < n) out[i] = (__bf16)in[i];
}

// ---------------- bf16 WMMA GEMM:  Out[M,N] = A[M,K] * B[N,K]^T + bias ----------------
// Block: 256 threads = 8 waves.  Tile: 256(M) x 64(N), K-step 32.
// Wave w owns M rows [w*32, w*32+32): 2 A fragments x 4 N sub-tiles = 8 WMMA accumulators;
// each B fragment is reused by both A fragments (8 WMMA per 12 ds_load_b128 per k-step).
template<bool STORE_BF16>
__global__ __launch_bounds__(256)
void gemm_wmma_bf16(const __bf16* __restrict__ A,
                    const __bf16* __restrict__ Bm,   // weights, N x K row-major
                    const float*  __restrict__ bias,
                    void*         __restrict__ Out,
                    int M, int N, int K)
{
    __shared__ alignas(16) __bf16 lA[256 * 32];   // 16 KB
    __shared__ alignas(16) __bf16 lB[64 * 32];    //  4 KB

    const int t     = threadIdx.x;
    const int wave  = t >> 5;
    const int lane  = t & 31;
    const int mBase = blockIdx.y * 256;
    const int nBase = blockIdx.x * 64;
    const int r     = lane & 15;
    const int half  = lane >> 4;

    f32x8 acc[2][4];
    #pragma unroll
    for (int ms = 0; ms < 2; ++ms)
        #pragma unroll
        for (int nt = 0; nt < 4; ++nt)
            #pragma unroll
            for (int i = 0; i < 8; ++i) acc[ms][nt][i] = 0.0f;

    for (int k0 = 0; k0 < K; k0 += 32) {
        // ---- stage A tile: 256 rows x 32 bf16 = 1024 x 16B (4 per thread) ----
        #pragma unroll
        for (int rep = 0; rep < 4; ++rep) {
            int j = t + rep * 256;
            int row = j >> 2, sub = j & 3;
            stage16(&A[(size_t)(mBase + row) * K + k0 + sub * 8],
                    &lA[row * 32 + sub * 8]);
        }
        // ---- stage B tile: 64 rows x 32 bf16 = 256 x 16B (1 per thread) ----
        {
            int row = t >> 2, sub = t & 3;
            stage16(&Bm[(size_t)(nBase + row) * K + k0 + sub * 8],
                    &lB[row * 32 + sub * 8]);
        }
        stage_wait();
        __syncthreads();

        // ---- A fragments (16x32, ISA 16-bit A layout), 2 M sub-tiles ----
        bf16x16 af[2];
        #pragma unroll
        for (int ms = 0; ms < 2; ++ms) {
            const int rowb = (wave * 32 + ms * 16 + r) * 32;
            #pragma unroll
            for (int e = 0; e < 8; ++e)
                af[ms][e] = lA[rowb + half * 8 + e];
            #pragma unroll
            for (int e = 0; e < 8; ++e)
                af[ms][8 + e] = lA[rowb + 16 + half * 8 + e];
        }

        // ---- 4 x (B fragment, 2 WMMA sharing it) ----
        #pragma unroll
        for (int nt = 0; nt < 4; ++nt) {
            bf16x16 bfv;
            #pragma unroll
            for (int e = 0; e < 16; ++e)   // B[k][n]: k = (e>>3)*16+half*8+(e&7), n = nt*16+r
                bfv[e] = lB[(nt * 16 + r) * 32 + (e >> 3) * 16 + half * 8 + (e & 7)];
            acc[0][nt] = __builtin_amdgcn_wmma_f32_16x16x32_bf16(
                false, af[0], false, bfv, (short)0, acc[0][nt], false, false);
            acc[1][nt] = __builtin_amdgcn_wmma_f32_16x16x32_bf16(
                false, af[1], false, bfv, (short)0, acc[1][nt], false, false);
        }
        __syncthreads();
    }

    // ---- store D (16x16 f32 C/D layout: VGPR i -> M = i + 8*half, N = lane&15) ----
    #pragma unroll
    for (int ms = 0; ms < 2; ++ms) {
        #pragma unroll
        for (int nt = 0; nt < 4; ++nt) {
            int n = nBase + nt * 16 + r;
            float bvv = bias ? bias[n] : 0.0f;
            #pragma unroll
            for (int i = 0; i < 8; ++i) {
                int m = mBase + wave * 32 + ms * 16 + half * 8 + i;
                float v = acc[ms][nt][i] + bvv;
                if (STORE_BF16) ((__bf16*)Out)[(size_t)m * N + n] = (__bf16)v;
                else            ((float*) Out)[(size_t)m * N + n] = v;
            }
        }
    }
}

// ---------------- softmax over 16 (per b,q,head) ----------------
__global__ void softmax16(float* __restrict__ a, int nrows) {
    int i = blockIdx.x * blockDim.x + threadIdx.x;
    if (i >= nrows) return;
    size_t base = (size_t)i * 16;
    float v[16];
    float mx = -3.0e38f;
    #pragma unroll
    for (int j = 0; j < 16; ++j) { v[j] = a[base + j]; mx = fmaxf(mx, v[j]); }
    float s = 0.0f;
    #pragma unroll
    for (int j = 0; j < 16; ++j) { v[j] = __expf(v[j] - mx); s += v[j]; }
    float inv = 1.0f / s;
    #pragma unroll
    for (int j = 0; j < 16; ++j) a[base + j] = v[j] * inv;
}

// ---------------- trilinear deformable sampling ----------------
// One wave per (b, q, head); lane = channel within the head (Dh = 32 = wave32).
__global__ __launch_bounds__(256)
void msda_sample(const float*  __restrict__ ref,    // (2,8192,4,3)
                 const float*  __restrict__ off,    // (2,8192,384)
                 const float*  __restrict__ aw,     // (2,8192,128) softmaxed
                 const __bf16* __restrict__ val,    // (2,42240,256) bf16
                 __bf16*       __restrict__ out)    // (2,8192,256) bf16
{
    const int tid  = blockIdx.x * blockDim.x + threadIdx.x;
    const int lane = tid & 31;
    const int wv   = tid >> 5;          // (b*8192+q)*8 + h
    const int h    = wv & 7;
    const int q    = (wv >> 3) & (LQ - 1);
    const int b    = wv >> 16;

    const size_t rb = (size_t)(b * LQ + q) * (NLVL * 3);
    const size_t ob = (size_t)(b * LQ + q) * (HEADS * NLVL * POINTS * 3);
    const size_t ab = (size_t)(b * LQ + q) * (HEADS * NLVL * POINTS);

    float acc = 0.0f;

    #pragma unroll
    for (int l = 0; l < NLVL; ++l) {
        constexpr int CD[NLVL] = {8, 8, 4, 4};
        constexpr int CH[NLVL] = {64, 32, 16, 8};
        constexpr int CW[NLVL] = {64, 32, 16, 8};
        constexpr int CS[NLVL] = {0, 32768, 40960, 41984};
        const int   iD = CD[l], iH = CH[l], iW = CW[l], st = CS[l];
        const float fD = (float)iD, fH = (float)iH, fW = (float)iW;

        const float rx = ref[rb + l * 3 + 0];
        const float ry = ref[rb + l * 3 + 1];
        const float rz = ref[rb + l * 3 + 2];

        for (int p = 0; p < POINTS; ++p) {
            const size_t o = ob + (size_t)((h * NLVL + l) * POINTS + p) * 3;
            // x = loc_x * W - 0.5 (algebraic collapse of grid_sample mapping)
            const float x = (rx + off[o + 0] / fW) * fW - 0.5f;
            const float y = (ry + off[o + 1] / fH) * fH - 0.5f;
            const float z = (rz + off[o + 2] / fD) * fD - 0.5f;
            const float wat = aw[ab + h * (NLVL * POINTS) + l * POINTS + p];

            const float x0f = floorf(x), y0f = floorf(y), z0f = floorf(z);
            const float fx = x - x0f, fy = y - y0f, fz = z - z0f;
            const int x0 = (int)x0f, y0 = (int)y0f, z0 = (int)z0f;

            #pragma unroll
            for (int dz = 0; dz < 2; ++dz) {
                const int   zc = z0 + dz;
                const float wz = dz ? fz : 1.0f - fz;
                #pragma unroll
                for (int dy = 0; dy < 2; ++dy) {
                    const int   yc = y0 + dy;
                    const float wy = dy ? fy : 1.0f - fy;
                    #pragma unroll
                    for (int dx = 0; dx < 2; ++dx) {
                        const int   xc = x0 + dx;
                        const float wx = dx ? fx : 1.0f - fx;
                        if (xc >= 0 && xc < iW && yc >= 0 && yc < iH && zc >= 0 && zc < iD) {
                            const int idx = (zc * iH + yc) * iW + xc;
                            const size_t vo =
                                ((size_t)(b * S_TOT + st + idx)) * DMODEL + h * 32 + lane;
                            acc += wat * (wz * wy * wx) * (float)val[vo];
                        }
                    }
                }
            }
        }
    }
    out[(size_t)(b * LQ + q) * DMODEL + h * 32 + lane] = (__bf16)acc;
}

// ---------------- launch ----------------
extern "C" void kernel_launch(void* const* d_in, const int* in_sizes, int n_in,
                              void* d_out, int out_size, void* d_ws, size_t ws_size,
                              hipStream_t stream) {
    const float* query = (const float*)d_in[0];
    const float* ref   = (const float*)d_in[1];
    const float* xin   = (const float*)d_in[2];
    const float* Wv    = (const float*)d_in[3];
    const float* bv    = (const float*)d_in[4];
    const float* Wsamp = (const float*)d_in[5];
    const float* bsamp = (const float*)d_in[6];
    const float* Wattn = (const float*)d_in[7];
    const float* battn = (const float*)d_in[8];
    const float* Wo    = (const float*)d_in[9];
    const float* bo    = (const float*)d_in[10];

    char* ws = (char*)d_ws;
    size_t cur = 0;
    auto alloc = [&](size_t bytes) -> void* {
        void* p = ws + cur;
        cur = (cur + bytes + 255) & ~(size_t)255;
        return p;
    };

    __bf16* qbf     = (__bf16*)alloc((size_t)MQ * KDIM * 2);
    __bf16* xbf     = (__bf16*)alloc((size_t)MV * KDIM * 2);
    __bf16* wvbf    = (__bf16*)alloc((size_t)256 * KDIM * 2);
    __bf16* wsampbf = (__bf16*)alloc((size_t)384 * KDIM * 2);
    __bf16* wattnbf = (__bf16*)alloc((size_t)128 * KDIM * 2);
    __bf16* wobf    = (__bf16*)alloc((size_t)256 * KDIM * 2);
    __bf16* valbf   = (__bf16*)alloc((size_t)MV * DMODEL * 2);
    float*  offf    = (float*) alloc((size_t)MQ * 384 * 4);
    float*  attnf   = (float*) alloc((size_t)MQ * 128 * 4);
    __bf16* sobf    = (__bf16*)alloc((size_t)MQ * DMODEL * 2);

    const int T = 256;
    auto cvt = [&](const float* src, __bf16* dst, size_t n) {
        cvt_f32_bf16<<<(unsigned)((n + T - 1) / T), T, 0, stream>>>(src, dst, n);
    };
    cvt(query, qbf,     (size_t)MQ * KDIM);
    cvt(xin,   xbf,     (size_t)MV * KDIM);
    cvt(Wv,    wvbf,    (size_t)256 * KDIM);
    cvt(Wsamp, wsampbf, (size_t)384 * KDIM);
    cvt(Wattn, wattnbf, (size_t)128 * KDIM);
    cvt(Wo,    wobf,    (size_t)256 * KDIM);

    // value = X @ Wv^T + bv  -> bf16 (stays resident in 192MB L2 for the gathers)
    gemm_wmma_bf16<true><<<dim3(256 / 64, MV / 256), 256, 0, stream>>>(
        xbf, wvbf, bv, valbf, MV, 256, KDIM);
    // sampling offsets = Q @ Wsamp^T + bsamp -> f32
    gemm_wmma_bf16<false><<<dim3(384 / 64, MQ / 256), 256, 0, stream>>>(
        qbf, wsampbf, bsamp, offf, MQ, 384, KDIM);
    // attn logits = Q @ Wattn^T + battn -> f32
    gemm_wmma_bf16<false><<<dim3(128 / 64, MQ / 256), 256, 0, stream>>>(
        qbf, wattnbf, battn, attnf, MQ, 128, KDIM);
    // softmax over L*P = 16 per (b,q,head)
    softmax16<<<(MQ * HEADS + T - 1) / T, T, 0, stream>>>(attnf, MQ * HEADS);
    // trilinear gather + weighted sum -> bf16
    msda_sample<<<(MQ * HEADS * 32) / T, T, 0, stream>>>(ref, offf, attnf, valbf, sobf);
    // out = S @ Wo^T + bo -> f32 d_out
    gemm_wmma_bf16<false><<<dim3(256 / 64, MQ / 256), 256, 0, stream>>>(
        sobf, wobf, bo, (float*)d_out, MQ, 256, KDIM);
}
